// InfoNCE_with_false_negative_elimination_31164282699864
// MI455X (gfx1250) — matrix operand order repaired and checked by reading, not compiled, not run on hardware
//
#include <hip/hip_runtime.h>
#include <hip/hip_bf16.h>
#include <math.h>

#define NROWS 8192
#define DDIM  256
#define KSEL  4095      // max(1, int(0.5 * (N-1)))
#define INV_T 10.0f     // 1 / temperature
#define BSTRIDE 40      // padded LDS row stride in halves (32 data + 8 pad)

typedef __attribute__((ext_vector_type(16))) __bf16 v16bf;
typedef __attribute__((ext_vector_type(8)))  float  v8f;
typedef __attribute__((ext_vector_type(4)))  int    v4i;

typedef __attribute__((address_space(1))) v4i* as1_v4i;
typedef __attribute__((address_space(3))) v4i* as3_v4i;

union Frag {
  v16bf v;
  uint4 u[2];
};

// --- async global->LDS copy (CDNA5 path), with sync fallback --------------
#if __has_builtin(__builtin_amdgcn_global_load_async_to_lds_b128) && \
    __has_builtin(__builtin_amdgcn_s_wait_asynccnt)
#define HAVE_ASYNC_LDS 1
#else
#define HAVE_ASYNC_LDS 0
#endif

__device__ __forceinline__ void cp16_g2l(const __bf16* g, __bf16* l) {
#if HAVE_ASYNC_LDS
  as1_v4i g1 = (as1_v4i)(uintptr_t)g;                 // global: flat == AS1
  as3_v4i l3 = (as3_v4i)(unsigned)(uintptr_t)l;       // LDS offset = low 32b
  __builtin_amdgcn_global_load_async_to_lds_b128(g1, l3, 0, 0);
#else
  *(uint4*)l = *(const uint4*)g;
#endif
}

#if HAVE_ASYNC_LDS
#define WAIT_ASYNC(n) __builtin_amdgcn_s_wait_asynccnt((short)(n))
#else
#define WAIT_ASYNC(n) ((void)0)
#endif

__device__ __forceinline__ float key_to_float(unsigned key) {
  unsigned b = (key & 0x80000000u) ? (key ^ 0x80000000u) : ~key;
  return __uint_as_float(b);
}

// ---------------------------------------------------------------------------
// Kernel 1: L2-normalize rows, split fp32 -> bf16 hi + bf16 lo (bf16x3 prep)
// ---------------------------------------------------------------------------
__global__ void __launch_bounds__(DDIM)
normalize_split_kernel(const float* __restrict__ X,
                       __bf16* __restrict__ Xh,
                       __bf16* __restrict__ Xl) {
  __shared__ float red[DDIM];
  const int row = blockIdx.x;
  const int t   = threadIdx.x;
  float x = X[(size_t)row * DDIM + t];
  red[t] = x * x;
  __syncthreads();
  for (int s = DDIM / 2; s > 0; s >>= 1) {
    if (t < s) red[t] += red[t + s];
    __syncthreads();
  }
  float inv = 1.0f / sqrtf(red[0]);
  float xn  = x * inv;
  __bf16 hi = (__bf16)xn;
  float  lo = xn - (float)hi;
  Xh[(size_t)row * DDIM + t] = hi;
  Xl[(size_t)row * DDIM + t] = (__bf16)lo;
}

// ---------------------------------------------------------------------------
// Kernel 2: logits = Qn @ Pn^T via bf16x3 WMMA (f32 accumulate).
// Block = 256 threads (8 waves), 128x128 C tile.
// B panel (hi+lo, 128 rows x 32 halves) staged in LDS per k-step,
// double-buffered via async global->LDS loads (ASYNCcnt) when available.
// ---------------------------------------------------------------------------
__global__ void __launch_bounds__(256)
gemm_bf16x3_kernel(const __bf16* __restrict__ Qh, const __bf16* __restrict__ Ql,
                   const __bf16* __restrict__ Ph, const __bf16* __restrict__ Pl,
                   float* __restrict__ C) {
  __shared__ __bf16 BhS[2][128 * BSTRIDE];  // 2 x 10 KB
  __shared__ __bf16 BlS[2][128 * BSTRIDE];  // 2 x 10 KB

  const int wave = threadIdx.x >> 5;
  const int lane = threadIdx.x & 31;
  const int half = lane >> 4;
  const int r    = lane & 15;
  const int m0   = blockIdx.y * 128 + wave * 16;
  const int n0   = blockIdx.x * 128;

  // --- staging assignment: thread -> (row, 16-half segment) of B panel ---
  const int rowL = threadIdx.x >> 1;        // 0..127
  const int segH = (threadIdx.x & 1) * 16;  // 0 or 16 halves
  const __bf16* gBh = Ph + (size_t)(n0 + rowL) * DDIM + segH;
  const __bf16* gBl = Pl + (size_t)(n0 + rowL) * DDIM + segH;
  __bf16* lBh = &BhS[0][rowL * BSTRIDE + segH];
  __bf16* lBl = &BlS[0][rowL * BSTRIDE + segH];
  const int lbuf = 128 * BSTRIDE;

  v8f acc[8];
#pragma unroll
  for (int t = 0; t < 8; ++t)
    acc[t] = (v8f){0.f, 0.f, 0.f, 0.f, 0.f, 0.f, 0.f, 0.f};

  const __bf16* aRowH = Qh + (size_t)(m0 + r) * DDIM;
  const __bf16* aRowL = Ql + (size_t)(m0 + r) * DDIM;

  // prologue: stage k-block 0 into buffer 0 (4 async loads per thread)
  cp16_g2l(gBh, lBh);
  cp16_g2l(gBh + 8, lBh + 8);
  cp16_g2l(gBl, lBl);
  cp16_g2l(gBl + 8, lBl + 8);

  for (int ks = 0; ks < 8; ++ks) {
    const int kb  = ks * 32;
    const int buf = ks & 1;

    if (ks < 7) {  // stage next k-block into the other buffer
      const int kn = kb + 32;
      __bf16* lh = lBh + (buf ^ 1) * lbuf;
      __bf16* ll = lBl + (buf ^ 1) * lbuf;
      cp16_g2l(gBh + kn, lh);
      cp16_g2l(gBh + kn + 8, lh + 8);
      cp16_g2l(gBl + kn, ll);
      cp16_g2l(gBl + kn + 8, ll + 8);
      WAIT_ASYNC(4);  // in-order: the 4 older (current-buffer) loads landed
    } else {
      WAIT_ASYNC(0);
    }
    __syncthreads();

    // A fragments direct from global (own 16 rows; L2-resident across blocks)
    Frag ah, al;
    const int ka = kb + half * 16;
    ah.u[0] = *(const uint4*)(aRowH + ka);
    ah.u[1] = *(const uint4*)(aRowH + ka + 8);
    al.u[0] = *(const uint4*)(aRowL + ka);
    al.u[1] = *(const uint4*)(aRowL + ka + 8);

    const int bOff = half * 16;  // halves within staged 32-half block
#pragma unroll
    for (int t = 0; t < 8; ++t) {
      const __bf16* bh_p = &BhS[buf][(t * 16 + r) * BSTRIDE + bOff];
      const __bf16* bl_p = &BlS[buf][(t * 16 + r) * BSTRIDE + bOff];
      Frag bh, bl;
      bh.u[0] = *(const uint4*)(bh_p);
      bh.u[1] = *(const uint4*)(bh_p + 8);
      bl.u[0] = *(const uint4*)(bl_p);
      bl.u[1] = *(const uint4*)(bl_p + 8);
      acc[t] = __builtin_amdgcn_wmma_f32_16x16x32_bf16(
          false, ah.v, false, bh.v, (short)0, acc[t], false, false);
      acc[t] = __builtin_amdgcn_wmma_f32_16x16x32_bf16(
          false, ah.v, false, bl.v, (short)0, acc[t], false, false);
      acc[t] = __builtin_amdgcn_wmma_f32_16x16x32_bf16(
          false, al.v, false, bh.v, (short)0, acc[t], false, false);
    }
    __syncthreads();  // all waves done reading buf before it is re-staged
  }

  // C/D layout: lane l, VGPR v -> row = m0 + (l/16)*8 + v, col = n0t + l%16
#pragma unroll
  for (int t = 0; t < 8; ++t) {
    const int col = n0 + t * 16 + r;
#pragma unroll
    for (int v = 0; v < 8; ++v) {
      const int rowm = m0 + half * 8 + v;
      C[(size_t)rowm * NROWS + col] = acc[t][v];
    }
  }
}

// ---------------------------------------------------------------------------
// Kernel 3: per-row exact radix-select of the 4095-th smallest negative logit
// (diagonal excluded), then sum-exp over the k smallest + positive -> loss_i.
// ---------------------------------------------------------------------------
__global__ void __launch_bounds__(256)
row_loss_kernel(const float* __restrict__ C, float* __restrict__ loss) {
  __shared__ unsigned keys[NROWS];
  __shared__ unsigned hist[256];
  __shared__ unsigned sel, selwant;
  __shared__ float    fred[256];
  __shared__ unsigned ired[256];

  const int row = blockIdx.x;
  const int t   = threadIdx.x;
  const float scale = INV_T * 1.44269504088896340736f;  // log2(e)/T

  for (int i = t; i < NROWS; i += 256) {
    float v = C[(size_t)row * NROWS + i];
    unsigned b = __float_as_uint(v);
    unsigned key = (b & 0x80000000u) ? ~b : (b | 0x80000000u);
    if (i == row) key = 0xFFFFFFFFu;
    keys[i] = key;
  }
  __syncthreads();

  unsigned prefix = 0;
  unsigned want   = KSEL;
  for (int shift = 24; shift >= 0; shift -= 8) {
    hist[t] = 0;
    __syncthreads();
    const unsigned pmask = (shift == 24) ? 0u : (0xFFFFFFFFu << (shift + 8));
    for (int i = t; i < NROWS; i += 256) {
      unsigned key = keys[i];
      if ((key & pmask) == prefix)
        atomicAdd(&hist[(key >> shift) & 0xFFu], 1u);
    }
    __syncthreads();
    if (t == 0) {
      unsigned cum = 0;
      for (int b = 0; b < 256; ++b) {
        unsigned c = hist[b];
        if (cum + c >= want) { sel = (unsigned)b; selwant = want - cum; break; }
        cum += c;
      }
    }
    __syncthreads();
    prefix |= sel << shift;
    want = selwant;
    __syncthreads();
  }

  const unsigned kthkey = prefix;
  const float    kthval = key_to_float(kthkey);

  float    partial = 0.f;
  unsigned cntless = 0;
  for (int i = t; i < NROWS; i += 256) {
    unsigned key = keys[i];
    if (key < kthkey) {
      partial += exp2f(key_to_float(key) * scale);
      ++cntless;
    }
  }
  fred[t] = partial;
  ired[t] = cntless;
  __syncthreads();
  for (int s = 128; s > 0; s >>= 1) {
    if (t < s) { fred[t] += fred[t + s]; ired[t] += ired[t + s]; }
    __syncthreads();
  }
  if (t == 0) {
    float sumneg = fred[0] + (float)(KSEL - ired[0]) * exp2f(kthval * scale);
    float pos = C[(size_t)row * NROWS + row];
    float S = sumneg + exp2f(pos * scale);
    loss[row] = logf(S) - pos * INV_T;
  }
}

// ---------------------------------------------------------------------------
// Kernel 4: mean over the 8192 per-row losses -> scalar output.
// ---------------------------------------------------------------------------
__global__ void __launch_bounds__(256)
mean_kernel(const float* __restrict__ loss, float* __restrict__ out) {
  __shared__ float red[256];
  const int t = threadIdx.x;
  float s = 0.f;
  for (int i = t; i < NROWS; i += 256) s += loss[i];
  red[t] = s;
  __syncthreads();
  for (int stp = 128; stp > 0; stp >>= 1) {
    if (t < stp) red[t] += red[t + stp];
    __syncthreads();
  }
  if (t == 0) out[0] = red[0] / (float)NROWS;
}

// ---------------------------------------------------------------------------
extern "C" void kernel_launch(void* const* d_in, const int* in_sizes, int n_in,
                              void* d_out, int out_size, void* d_ws, size_t ws_size,
                              hipStream_t stream) {
  (void)in_sizes; (void)n_in; (void)out_size; (void)ws_size;
  const float* q = (const float*)d_in[0];
  const float* p = (const float*)d_in[1];

  char* ws = (char*)d_ws;
  const size_t mat_elems = (size_t)NROWS * DDIM;
  __bf16* qh = (__bf16*)ws;
  __bf16* ql = qh + mat_elems;
  __bf16* ph = ql + mat_elems;
  __bf16* pl = ph + mat_elems;
  float* logits  = (float*)(pl + mat_elems);
  float* lossArr = logits + (size_t)NROWS * NROWS;
  float* out = (float*)d_out;

  normalize_split_kernel<<<NROWS, DDIM, 0, stream>>>(q, qh, ql);
  normalize_split_kernel<<<NROWS, DDIM, 0, stream>>>(p, ph, pl);

  dim3 grid(NROWS / 128, NROWS / 128);
  gemm_bf16x3_kernel<<<grid, 256, 0, stream>>>(qh, ql, ph, pl, logits);

  row_loss_kernel<<<NROWS, 256, 0, stream>>>(logits, lossArr);
  mean_kernel<<<1, 256, 0, stream>>>(lossArr, out);
}
